// MHAttention_1589137900296
// MI455X (gfx1250) — compile-verified, matrix operand order checked by reading, not run on hardware
//
#include <hip/hip_runtime.h>
#include <hip/hip_bf16.h>

#define D_MODEL 1024
#define NHID 64
#define NHEAD 16
#define BB 2
#define LL 2048

typedef __attribute__((ext_vector_type(16))) _Float16 v16h;
typedef __attribute__((ext_vector_type(8)))  _Float16 v8h;
typedef __attribute__((ext_vector_type(8)))  float    v8f;
typedef __attribute__((ext_vector_type(4)))  unsigned int v4u;
typedef __attribute__((ext_vector_type(8)))  int v8i;
typedef __attribute__((ext_vector_type(4)))  int v4i;

// ---------------------------------------------------------------------------
// D = A(16x32 f16) x B(32x16 f16) + C(16x16 f32), wave32 WMMA
// ---------------------------------------------------------------------------
__device__ __forceinline__ v8f wmma16(v16h a, v16h b, v8f c) {
  return __builtin_amdgcn_wmma_f32_16x16x32_f16(
      /*neg_a=*/false, a, /*neg_b=*/false, b,
      /*c_mod=*/(short)0, c, /*reuse_a=*/false, /*reuse_b=*/false);
}

// A-fragment (16 rows x 32 K) from a row-major f16 matrix, ld = row stride.
// ISA 7.12.2 16-bit A layout: lane = M row (mod 16); per lane two contiguous
// 8-halfword runs at K = koff..+8 and 16+koff..+8, koff=(lane>>4)*8.
// Doubles as a B-fragment when "B" is stored N x K row-major.
__device__ __forceinline__ v16h ldfrag_rm(const _Float16* __restrict__ base,
                                          int ld, int r0, int k0) {
  const int lane = threadIdx.x & 31;
  const int m    = lane & 15;
  const int koff = (lane >> 4) << 3;
  const _Float16* p = base + (size_t)(r0 + m) * ld + (k0 + koff);
  v8h lo = *(const v8h*)p;
  v8h hi = *(const v8h*)(p + 16);
  v16h f;
#pragma unroll
  for (int i = 0; i < 8; ++i) { f[i] = lo[i]; f[i + 8] = hi[i]; }
  return f;
}

// Same fragment pattern but from an LDS-resident 16-row x 32-K slab (ld = 32).
__device__ __forceinline__ v16h ldfrag_lds(const _Float16* base, int r0) {
  const int lane = threadIdx.x & 31;
  const int m    = lane & 15;
  const int koff = (lane >> 4) << 3;
  const _Float16* p = base + (r0 + m) * 32 + koff;
  v8h lo = *(const v8h*)p;          // ds_load_b128
  v8h hi = *(const v8h*)(p + 16);   // ds_load_b128
  v16h f;
#pragma unroll
  for (int i = 0; i < 8; ++i) { f[i] = lo[i]; f[i + 8] = hi[i]; }
  return f;
}

// B-fragment (32 K x 16 N) from a K-major (row-major K x N) f16 matrix.
__device__ __forceinline__ v16h ldfrag_cm(const _Float16* __restrict__ base,
                                          int ld, int k0, int n0) {
  const int lane = threadIdx.x & 31;
  const int n    = lane & 15;
  const int koff = (lane >> 4) << 3;
  const _Float16* p = base + (size_t)(k0 + koff) * ld + (n0 + n);
  v16h f;
#pragma unroll
  for (int i = 0; i < 8; ++i) {
    f[i]     = p[(size_t)i * ld];
    f[i + 8] = p[(size_t)(16 + i) * ld];
  }
  return f;
}

// ---------------------------------------------------------------------------
// TDM: DMA a 128-row x 32-K f16 tile of A[M x K] (row-major) into LDS.
// D# per CDNA5 ISA: group0 = {count=1, lds_addr, global_addr, type=2},
// group1 = {wg_mask=0, data_size=2B, tensor_dim0=K, tensor_dim1=M,
//           tile_dim0=32, tile_dim1=128, dim0_stride=K}.
// This toolchain exposes the 6-arg builtin (g0, g1, g2, g3, extra, cpol).
// ---------------------------------------------------------------------------
__device__ __forceinline__ void tdm_load_a_tile(const _Float16* gtile,
                                                unsigned lds_byte_addr,
                                                int K, int M) {
  const unsigned long long ga = (unsigned long long)(uintptr_t)gtile;
  v4u g0;
  g0[0] = 1u;                                        // count=1, user mode
  g0[1] = lds_byte_addr;                             // LDS dest (bytes)
  g0[2] = (unsigned)(ga & 0xffffffffu);              // global_addr[31:0]
  g0[3] = (unsigned)((ga >> 32) & 0x01ffffffu) | (2u << 30);  // addr[56:32]|type=2
  v8i g1;
  g1[0] = (int)(1u << 16);                           // wg_mask=0, data_size=2B
  g1[1] = (int)(((unsigned)K & 0xffffu) << 16);      // tensor_dim0 lo16
  g1[2] = (int)((((unsigned)K >> 16) & 0xffffu) | (((unsigned)M & 0xffffu) << 16));
  g1[3] = (int)((((unsigned)M >> 16) & 0xffffu) | (32u << 16));  // tile_dim0=32
  g1[4] = 128;                                       // tile_dim1=128, tile_dim2=0
  g1[5] = K;                                         // dim0_stride lo32
  g1[6] = 0;                                         // dim0_stride hi | dim1_stride lo
  g1[7] = 0;
  v4i z4 = {0, 0, 0, 0};
  v8i z8 = {0, 0, 0, 0, 0, 0, 0, 0};
  __builtin_amdgcn_tensor_load_to_lds(g0, g1, z4, z4, z8, /*cpol=*/0);
}

// ---------------------------------------------------------------------------
// f32 -> f16 conversion
// ---------------------------------------------------------------------------
__global__ void f32_to_f16_kernel(const float* __restrict__ in,
                                  _Float16* __restrict__ out, int n) {
  int i = blockIdx.x * blockDim.x + threadIdx.x;
  if (i < n) out[i] = (_Float16)in[i];
}

// ---------------------------------------------------------------------------
// C[M,N] = A[M,K] @ W[N,K]^T (torch Linear convention, f16 in, f32 accum).
// Block: 256 threads = 8 waves (2 M x 4 N), block tile 128x128, wave 64x32.
// A block-slab (128x32) is TDM-DMA'd into LDS, double-buffered; wave0 issues
// the tensor op for step i+1 while all waves consume step i. B comes straight
// from global (weights are L2-resident: 192 MB L2 >> 2 MB per matrix).
// mode 0: scatter f16 into head-major [b, h, l, 64] (packed col c -> h=c%16,
//         d=c/16 because hidden is the OUTER factor in the reference reshape)
// mode 1: dense f32 row-major [M, N]
// ---------------------------------------------------------------------------
__global__ __launch_bounds__(256) void gemm_wmma(
    const _Float16* __restrict__ A, const _Float16* __restrict__ W,
    void* __restrict__ out, int M, int N, int K, int mode) {
  __shared__ __align__(16) _Float16 ldsA[2][128 * 32];
  const int wave = threadIdx.x >> 5;
  const int bm0 = blockIdx.y * 128;
  const int m0w = (wave >> 2) * 64;                       // wave row inside block
  const int n0  = blockIdx.x * 128 + (wave & 3) * 32;
  const int steps = K >> 5;
  const unsigned lds0 = (unsigned)(unsigned long long)(uintptr_t)(&ldsA[0][0]);
  const unsigned ldsStride = 128 * 32 * sizeof(_Float16);

  if (wave == 0)
    tdm_load_a_tile(A + (size_t)bm0 * K, lds0, K, M);

  v8f acc[4][2] = {};
  for (int i = 0; i < steps; ++i) {
    const int k0 = i << 5;
    if (wave == 0) {
      if (i + 1 < steps) {
        tdm_load_a_tile(A + (size_t)bm0 * K + (k0 + 32),
                        lds0 + (unsigned)((i + 1) & 1) * ldsStride, K, M);
        __builtin_amdgcn_s_wait_tensorcnt(1);   // current buf's DMA done
      } else {
        __builtin_amdgcn_s_wait_tensorcnt(0);   // last step: drain
      }
    }
    __syncthreads();                    // buf[i&1] is ready for everyone
    const _Float16* As = ldsA[i & 1];
    v16h af[4], bf[2];
#pragma unroll
    for (int t = 0; t < 4; ++t) af[t] = ldfrag_lds(As, m0w + 16 * t);
#pragma unroll
    for (int t = 0; t < 2; ++t) bf[t] = ldfrag_rm(W, K, n0 + 16 * t, k0);
    if (k0 + 32 < K)  // nudge next W slab toward L2 (global_prefetch_b8)
      __builtin_prefetch(W + (size_t)(n0 + (threadIdx.x & 31)) * K + k0 + 32, 0, 1);
#pragma unroll
    for (int ti = 0; ti < 4; ++ti)
#pragma unroll
      for (int tj = 0; tj < 2; ++tj)
        acc[ti][tj] = wmma16(af[ti], bf[tj], acc[ti][tj]);
    __syncthreads();                    // all reads of buf[i&1] done before refill
  }

  const int lane = threadIdx.x & 31;
  const int cn = lane & 15;
  const int rb = (lane >> 4) << 3;
#pragma unroll
  for (int ti = 0; ti < 4; ++ti)
#pragma unroll
    for (int tj = 0; tj < 2; ++tj)
#pragma unroll
      for (int r = 0; r < 8; ++r) {
        const int gm = bm0 + m0w + 16 * ti + rb + r;
        const int gn = n0 + 16 * tj + cn;
        const float val = acc[ti][tj][r];
        if (mode == 0) {
          const int bb = gm / LL, l = gm % LL;
          const int h = gn & 15, d = gn >> 4;
          ((_Float16*)out)[(((size_t)(bb * NHEAD + h)) * LL + l) * NHID + d] =
              (_Float16)val;
        } else {
          ((float*)out)[(size_t)gm * N + gn] = val;
        }
      }
}

// ---------------------------------------------------------------------------
// Flash attention per (b*16+h, query block of 64). 128 threads = 4 waves,
// wave owns 16 query rows x full d=64. Online softmax in f32; P staged
// through LDS to convert C-fragment layout into an A-fragment.
// ---------------------------------------------------------------------------
__global__ __launch_bounds__(128) void attn_wmma(
    const _Float16* __restrict__ Qh, const _Float16* __restrict__ Kh,
    const _Float16* __restrict__ Vh, _Float16* __restrict__ Oout) {
  __shared__ __align__(16) _Float16 ldsP[4][16 * 32];
  const int wave = threadIdx.x >> 5;
  const int lane = threadIdx.x & 31;
  const int bh = blockIdx.x;                 // b*16 + h
  const int r0 = blockIdx.y * 64 + wave * 16;

  const _Float16* Q  = Qh + (size_t)bh * LL * NHID;
  const _Float16* Kx = Kh + (size_t)bh * LL * NHID;
  const _Float16* V  = Vh + (size_t)bh * LL * NHID;

  const v16h qa0 = ldfrag_rm(Q, NHID, r0, 0);
  const v16h qa1 = ldfrag_rm(Q, NHID, r0, 32);

  v8f o[4] = {};
  float mrow[8], lrow[8];
#pragma unroll
  for (int r = 0; r < 8; ++r) { mrow[r] = -1e30f; lrow[r] = 0.0f; }
  const float scale = 0.125f;  // 1/sqrt(64)

  for (int j0 = 0; j0 < LL; j0 += 32) {
    v8f s[2] = {};
#pragma unroll
    for (int t = 0; t < 2; ++t) {
      v16h kb0 = ldfrag_rm(Kx, NHID, j0 + 16 * t, 0);
      v16h kb1 = ldfrag_rm(Kx, NHID, j0 + 16 * t, 32);
      s[t] = wmma16(qa0, kb0, s[t]);
      s[t] = wmma16(qa1, kb1, s[t]);
    }
    float rmax[8];
#pragma unroll
    for (int r = 0; r < 8; ++r) {
      s[0][r] *= scale; s[1][r] *= scale;
      rmax[r] = fmaxf(s[0][r], s[1][r]);
    }
#pragma unroll
    for (int mask = 1; mask < 16; mask <<= 1)
#pragma unroll
      for (int r = 0; r < 8; ++r)
        rmax[r] = fmaxf(rmax[r], __shfl_xor(rmax[r], mask, 32));

    float corr[8], rsum[8];
#pragma unroll
    for (int r = 0; r < 8; ++r) {
      const float mnew = fmaxf(mrow[r], rmax[r]);
      corr[r] = __expf(mrow[r] - mnew);
      mrow[r] = mnew;
      rsum[r] = 0.0f;
    }
#pragma unroll
    for (int t = 0; t < 2; ++t)
#pragma unroll
      for (int r = 0; r < 8; ++r) {
        const float p = __expf(s[t][r] - mrow[r]);
        s[t][r] = p;
        rsum[r] += p;
      }
#pragma unroll
    for (int mask = 1; mask < 16; mask <<= 1)
#pragma unroll
      for (int r = 0; r < 8; ++r)
        rsum[r] += __shfl_xor(rsum[r], mask, 32);
#pragma unroll
    for (int r = 0; r < 8; ++r)
      lrow[r] = lrow[r] * corr[r] + rsum[r];
#pragma unroll
    for (int t = 0; t < 4; ++t)
#pragma unroll
      for (int r = 0; r < 8; ++r)
        o[t][r] *= corr[r];

    _Float16* Pl = ldsP[wave];
    {
      const int cn = lane & 15;
      const int rb = (lane >> 4) << 3;
#pragma unroll
      for (int t = 0; t < 2; ++t)
#pragma unroll
        for (int r = 0; r < 8; ++r)
          Pl[(rb + r) * 32 + 16 * t + cn] = (_Float16)s[t][r];
    }
    asm volatile("s_wait_dscnt 0x0" ::: "memory");
    v16h pa;
    {
      const int m = lane & 15;
      const int koff = (lane >> 4) << 3;
      const _Float16* p = Pl + m * 32 + koff;
#pragma unroll
      for (int i = 0; i < 8; ++i) { pa[i] = p[i]; pa[i + 8] = p[16 + i]; }
    }
#pragma unroll
    for (int t = 0; t < 4; ++t) {
      v16h vb = ldfrag_cm(V, NHID, j0, 16 * t);
      o[t] = wmma16(pa, vb, o[t]);
    }
  }

  const int cn = lane & 15;
  const int rb = (lane >> 4) << 3;
  const int bb = bh >> 4;
  const int h  = bh & 15;
#pragma unroll
  for (int t = 0; t < 4; ++t)
#pragma unroll
    for (int r = 0; r < 8; ++r) {
      const int l = r0 + rb + r;
      const int d = 16 * t + cn;
      Oout[((size_t)(bb * LL + l)) * (NHID * NHEAD) + d * NHEAD + h] =
          (_Float16)(o[t][r] / lrow[r]);
    }
}

// ---------------------------------------------------------------------------
// Host launch: convert -> 3x proj GEMM -> flash attention -> output GEMM
// ---------------------------------------------------------------------------
extern "C" void kernel_launch(void* const* d_in, const int* in_sizes, int n_in,
                              void* d_out, int out_size, void* d_ws, size_t ws_size,
                              hipStream_t stream) {
  (void)in_sizes; (void)n_in; (void)out_size; (void)ws_size;
  const float* q  = (const float*)d_in[0];
  const float* k  = (const float*)d_in[1];
  const float* v  = (const float*)d_in[2];
  const float* Wq = (const float*)d_in[3];
  const float* Wk = (const float*)d_in[4];
  const float* Wv = (const float*)d_in[5];
  const float* Wo = (const float*)d_in[6];

  const int M = BB * LL;                         // 4096 rows
  const size_t actN = (size_t)M * D_MODEL;       // 4 Mi elems
  const size_t wN   = (size_t)D_MODEL * D_MODEL; // 1 Mi elems

  _Float16* qf  = (_Float16*)d_ws;
  _Float16* kf  = qf  + actN;
  _Float16* vf  = kf  + actN;
  _Float16* wqf = vf  + actN;
  _Float16* wkf = wqf + wN;
  _Float16* wvf = wkf + wN;
  _Float16* wof = wvf + wN;
  _Float16* Qhd = wof + wN;
  _Float16* Khd = Qhd + actN;
  _Float16* Vhd = Khd + actN;
  _Float16* Ao  = qf;  // reuse q-activation buffer for attention output

  const int TB = 256;
  f32_to_f16_kernel<<<dim3((unsigned)((actN + TB - 1) / TB)), TB, 0, stream>>>(q, qf, (int)actN);
  f32_to_f16_kernel<<<dim3((unsigned)((actN + TB - 1) / TB)), TB, 0, stream>>>(k, kf, (int)actN);
  f32_to_f16_kernel<<<dim3((unsigned)((actN + TB - 1) / TB)), TB, 0, stream>>>(v, vf, (int)actN);
  f32_to_f16_kernel<<<dim3((unsigned)((wN + TB - 1) / TB)), TB, 0, stream>>>(Wq, wqf, (int)wN);
  f32_to_f16_kernel<<<dim3((unsigned)((wN + TB - 1) / TB)), TB, 0, stream>>>(Wk, wkf, (int)wN);
  f32_to_f16_kernel<<<dim3((unsigned)((wN + TB - 1) / TB)), TB, 0, stream>>>(Wv, wvf, (int)wN);
  f32_to_f16_kernel<<<dim3((unsigned)((wN + TB - 1) / TB)), TB, 0, stream>>>(Wo, wof, (int)wN);

  const dim3 gg(D_MODEL / 128, M / 128);  // (8, 32)
  gemm_wmma<<<gg, 256, 0, stream>>>(qf, wqf, Qhd, M, D_MODEL, D_MODEL, 0);
  gemm_wmma<<<gg, 256, 0, stream>>>(kf, wkf, Khd, M, D_MODEL, D_MODEL, 0);
  gemm_wmma<<<gg, 256, 0, stream>>>(vf, wvf, Vhd, M, D_MODEL, D_MODEL, 0);

  attn_wmma<<<dim3(BB * NHEAD, LL / 64), 128, 0, stream>>>(Qhd, Khd, Vhd, Ao);

  gemm_wmma<<<gg, 256, 0, stream>>>(Ao, wof, d_out, M, D_MODEL, D_MODEL, 1);
}
